// EdgeLevelAttentionLayer_27934467293580
// MI455X (gfx1250) — matrix operand order repaired and checked by reading, not compiled, not run on hardware
//
#include <hip/hip_runtime.h>
#include <stdint.h>

// Problem constants (match reference)
#define N_NODES   2048
#define N_EDGES   4096
#define FEAT      256
#define EMB       64

typedef __attribute__((ext_vector_type(16))) __bf16 v16bf;
typedef __attribute__((ext_vector_type(8)))  float  v8f;

__device__ __forceinline__ unsigned short f32_to_bf16(float x) {
    unsigned int u = __float_as_uint(x);
    u += 0x7FFFu + ((u >> 16) & 1u);       // round-to-nearest-even
    return (unsigned short)(u >> 16);
}

__device__ __forceinline__ float leaky01(float x) {
    return x > 0.0f ? x : 0.01f * x;
}

// ---------------------------------------------------------------------------
// Kernel 1: embeds (X @ W), score dot-products, transposed bf16 embeds.
// One block of 64 threads per output row (node rows first, then edge rows).
// Thread n owns embedding column n.
// ---------------------------------------------------------------------------
__global__ __launch_bounds__(64)
void gat_precompute_kernel(const float* __restrict__ node_features,
                           const float* __restrict__ edge_features,
                           const float* __restrict__ weight_node,
                           const float* __restrict__ weight_edge,
                           const float* __restrict__ pv_node,   // [128]
                           const float* __restrict__ pv_edge,   // [128]
                           unsigned short* __restrict__ embT_node, // [64][N_NODES] bf16
                           unsigned short* __restrict__ embT_edge, // [64][N_EDGES] bf16
                           float* __restrict__ a_e,   // [E] edge_emb . pv_node[:64]
                           float* __restrict__ b_n,   // [N] node_emb . pv_node[64:]
                           float* __restrict__ c_e,   // [E] edge_emb . pv_edge[:64]
                           float* __restrict__ d2_e)  // [E] edge_emb . pv_edge[64:]
{
    __shared__ float red0[64], red1[64], red2[64];
    const int row = blockIdx.x;
    const int n   = threadIdx.x;          // 0..63

    if (row < N_NODES) {
        const float* f = node_features + (size_t)row * FEAT;
        float s = 0.0f;
        #pragma unroll 4
        for (int k = 0; k < FEAT; ++k) s = fmaf(f[k], weight_node[k * EMB + n], s);
        embT_node[(size_t)n * N_NODES + row] = f32_to_bf16(s);
        red0[n] = s * pv_node[EMB + n];
        __syncthreads();
        for (int off = 32; off > 0; off >>= 1) {
            if (n < off) red0[n] += red0[n + off];
            __syncthreads();
        }
        if (n == 0) b_n[row] = red0[0];
    } else {
        const int e = row - N_NODES;
        const float* f = edge_features + (size_t)e * FEAT;
        float s = 0.0f;
        #pragma unroll 4
        for (int k = 0; k < FEAT; ++k) s = fmaf(f[k], weight_edge[k * EMB + n], s);
        embT_edge[(size_t)n * N_EDGES + e] = f32_to_bf16(s);
        red0[n] = s * pv_node[n];
        red1[n] = s * pv_edge[n];
        red2[n] = s * pv_edge[EMB + n];
        __syncthreads();
        for (int off = 32; off > 0; off >>= 1) {
            if (n < off) {
                red0[n] += red0[n + off];
                red1[n] += red1[n + off];
                red2[n] += red2[n + off];
            }
            __syncthreads();
        }
        if (n == 0) { a_e[e] = red0[0]; c_e[e] = red1[0]; d2_e[e] = red2[0]; }
    }
}

// ---------------------------------------------------------------------------
// Kernel 2: fused attention.
//   pass 0: node-to-edge  (K = N_NODES, writes out cols [0,64))
//   pass 1: edge-to-edge  (K = N_EDGES, writes out cols [64,128))
// Block = 256 threads = 8 wave32; block owns a 16-row output tile [16,64].
// Each wave strides K in 32-wide steps; per step it builds the 16x32 bf16
// A-fragment w[e,k] = adj * exp(leaky(a[e]+b[k])) in contiguous k order (k = 16*hf + j), same as B
// and issues 4x v_wmma_f32_16x16x32_bf16 against transposed bf16 embeds.
// ---------------------------------------------------------------------------
__global__ __launch_bounds__(256)
void gat_attention_kernel(const float* __restrict__ adj_n,   // [E][N_NODES]
                          const float* __restrict__ adj_e,   // [E][N_EDGES]
                          const unsigned short* __restrict__ embT_node,
                          const unsigned short* __restrict__ embT_edge,
                          const float* __restrict__ a_e,
                          const float* __restrict__ b_n,
                          const float* __restrict__ c_e,
                          const float* __restrict__ d2_e,
                          float* __restrict__ out)            // [E][128]
{
    __shared__ float lds_b[N_EDGES];       // column-score vector (<=16KB)
    __shared__ float lds_acc[16 * EMB];    // 16x64 f32 tile accumulator
    __shared__ float lds_rowsum[16];
    __shared__ float lds_denom[16];

    const int pass = blockIdx.y;
    const int tile = blockIdx.x;

    const int                   K    = (pass == 0) ? N_NODES : N_EDGES;
    const float* __restrict__   adj  = (pass == 0) ? adj_n    : adj_e;
    const float* __restrict__   avec = (pass == 0) ? a_e      : c_e;
    const float* __restrict__   bvec = (pass == 0) ? b_n      : d2_e;
    const unsigned short* __restrict__ embT = (pass == 0) ? embT_node : embT_edge;

    const int tid = threadIdx.x;

    for (int i = tid; i < K; i += 256)       lds_b[i]   = bvec[i];
    for (int i = tid; i < 16 * EMB; i += 256) lds_acc[i] = 0.0f;
    if (tid < 16) { lds_rowsum[tid] = 0.0f; lds_denom[tid] = 0.0f; }
    __syncthreads();

    const int wave = tid >> 5;
    const int lane = tid & 31;
    const int m    = lane & 15;     // output row within tile
    const int hf   = lane >> 4;     // wave half (A/B/C fragment layout)
    const int e_row = tile * 16 + m;

    const float  a_val   = avec[e_row];
    const float* adj_row = adj + (size_t)e_row * K;

    v8f acc0 = {}, acc1 = {}, acc2 = {}, acc3 = {};
    float rowsum = 0.0f, denom = 0.0f;

    union AFrag { v16bf v; unsigned short u[16]; };
    union BFrag { v16bf v; uint4 q[2]; };

    for (int kbase = wave * 32; kbase < K; kbase += 256) {
        // A-fragment k-indices (contiguous order, matching B):
        //   element j -> k = kbase + hf*16 + j
        const int b0 = kbase + hf * 16;
        const int b1 = b0 + 8;

        float adjv[16], bb[16];
        *(float4*)(adjv + 0)  = *(const float4*)(adj_row + b0);
        *(float4*)(adjv + 4)  = *(const float4*)(adj_row + b0 + 4);
        *(float4*)(adjv + 8)  = *(const float4*)(adj_row + b1);
        *(float4*)(adjv + 12) = *(const float4*)(adj_row + b1 + 4);
        *(float4*)(bb + 0)    = *(const float4*)(lds_b + b0);
        *(float4*)(bb + 4)    = *(const float4*)(lds_b + b0 + 4);
        *(float4*)(bb + 8)    = *(const float4*)(lds_b + b1);
        *(float4*)(bb + 12)   = *(const float4*)(lds_b + b1 + 4);

        AFrag af;
        #pragma unroll
        for (int j = 0; j < 16; ++j) {
            const float av = adjv[j];
            const float s  = leaky01(a_val + bb[j]);
            const float w  = (av != 0.0f) ? av * __expf(s) : 0.0f;
            rowsum += w;
            denom  += av;
            af.u[j] = f32_to_bf16(w);
        }

        // B-fragments: lane holds embT[n0+m][kbase + hf*16 + 0..15] (32B contiguous)
        const int klo = kbase + hf * 16;
        BFrag bf0, bf1, bf2, bf3;
        const uint4* p0 = (const uint4*)(embT + (size_t)( 0 + m) * K + klo);
        const uint4* p1 = (const uint4*)(embT + (size_t)(16 + m) * K + klo);
        const uint4* p2 = (const uint4*)(embT + (size_t)(32 + m) * K + klo);
        const uint4* p3 = (const uint4*)(embT + (size_t)(48 + m) * K + klo);
        bf0.q[0] = p0[0]; bf0.q[1] = p0[1];
        bf1.q[0] = p1[0]; bf1.q[1] = p1[1];
        bf2.q[0] = p2[0]; bf2.q[1] = p2[1];
        bf3.q[0] = p3[0]; bf3.q[1] = p3[1];

        acc0 = __builtin_amdgcn_wmma_f32_16x16x32_bf16(false, af.v, false, bf0.v, (short)0, acc0, false, false);
        acc1 = __builtin_amdgcn_wmma_f32_16x16x32_bf16(false, af.v, false, bf1.v, (short)0, acc1, false, false);
        acc2 = __builtin_amdgcn_wmma_f32_16x16x32_bf16(false, af.v, false, bf2.v, (short)0, acc2, false, false);
        acc3 = __builtin_amdgcn_wmma_f32_16x16x32_bf16(false, af.v, false, bf3.v, (short)0, acc3, false, false);
    }

    // Cross-wave reduction: D element r lives at row m = hf*8 + r, col = n0 + m
    #pragma unroll
    for (int r = 0; r < 8; ++r) {
        const int mrow = hf * 8 + r;
        atomicAdd(&lds_acc[mrow * EMB +  0 + m], acc0[r]);
        atomicAdd(&lds_acc[mrow * EMB + 16 + m], acc1[r]);
        atomicAdd(&lds_acc[mrow * EMB + 32 + m], acc2[r]);
        atomicAdd(&lds_acc[mrow * EMB + 48 + m], acc3[r]);
    }
    atomicAdd(&lds_rowsum[m], rowsum);
    atomicAdd(&lds_denom[m], denom);
    __syncthreads();

    // Finalize: val = acc / (rowsum * denom), nan_to_num, leaky, write concat cols
    const int colbase = pass * EMB;
    for (int idx = tid; idx < 16 * EMB; idx += 256) {
        const int mm = idx / EMB;
        const int nn = idx % EMB;
        float val = lds_acc[idx] / (lds_rowsum[mm] * lds_denom[mm]);
        if (isnan(val))       val = 0.0f;
        else if (isinf(val))  val = (val > 0.0f) ? 3.4028235e38f : -3.4028235e38f;
        out[(size_t)(tile * 16 + mm) * 128 + colbase + nn] = leaky01(val);
    }
}

// ---------------------------------------------------------------------------
// Host launcher
// ---------------------------------------------------------------------------
extern "C" void kernel_launch(void* const* d_in, const int* in_sizes, int n_in,
                              void* d_out, int out_size, void* d_ws, size_t ws_size,
                              hipStream_t stream) {
    (void)in_sizes; (void)n_in; (void)out_size; (void)ws_size;

    const float* node_features = (const float*)d_in[0];  // [2048,256]
    const float* edge_features = (const float*)d_in[1];  // [4096,256]
    const float* adj_e         = (const float*)d_in[2];  // [4096,4096]
    const float* adj_n         = (const float*)d_in[3];  // [4096,2048]
    const float* weight_node   = (const float*)d_in[4];  // [256,64]
    const float* weight_edge   = (const float*)d_in[5];  // [256,64]
    const float* pv_node       = (const float*)d_in[6];  // [128]
    const float* pv_edge       = (const float*)d_in[7];  // [128]
    float* out = (float*)d_out;                          // [4096,128]

    // Workspace layout (all offsets 256B-aligned), ~824KB total
    char* ws = (char*)d_ws;
    unsigned short* embT_node = (unsigned short*)(ws + 0);        // 64*2048*2 = 256KB
    unsigned short* embT_edge = (unsigned short*)(ws + 262144);   // 64*4096*2 = 512KB
    float* a_e  = (float*)(ws + 786432);                          // 16KB
    float* b_n  = (float*)(ws + 802816);                          // 8KB
    float* c_e  = (float*)(ws + 811008);                          // 16KB
    float* d2_e = (float*)(ws + 827392);                          // 16KB

    gat_precompute_kernel<<<dim3(N_NODES + N_EDGES), dim3(64), 0, stream>>>(
        node_features, edge_features, weight_node, weight_edge,
        pv_node, pv_edge, embT_node, embT_edge, a_e, b_n, c_e, d2_e);

    gat_attention_kernel<<<dim3(N_EDGES / 16, 2), dim3(256), 0, stream>>>(
        adj_n, adj_e, embT_node, embT_edge, a_e, b_n, c_e, d2_e, out);
}
